// SDTAEncoder_28973849379189
// MI455X (gfx1250) — matrix operand order repaired and checked by reading, not compiled, not run on hardware
//
#include <hip/hip_runtime.h>
#include <hip/hip_bf16.h>

typedef __attribute__((ext_vector_type(16))) _Float16 v16h;
typedef __attribute__((ext_vector_type(8)))  _Float16 v8h;
typedef __attribute__((ext_vector_type(8)))  float    v8f;

typedef __attribute__((ext_vector_type(4))) unsigned int tdm_g0_t;
typedef __attribute__((ext_vector_type(8))) int          tdm_g1_t;
typedef __attribute__((ext_vector_type(4))) int          tdm_g2_t;

#define DIMC   256
#define NTOK   1024
#define HEADS  8
#define HDIM   32
#define BATCH  8

#if defined(__has_builtin)
#if __has_builtin(__builtin_amdgcn_tensor_load_to_lds) && __has_builtin(__builtin_amdgcn_s_wait_tensorcnt)
#define HAVE_TDM 1
#endif
#endif
#ifndef HAVE_TDM
#define HAVE_TDM 0
#endif

// ---------------------------------------------------------------------------
// Helpers
// ---------------------------------------------------------------------------
__device__ __forceinline__ v16h cat8(v8h lo, v8h hi) {
    return __builtin_shufflevector(lo, hi, 0,1,2,3,4,5,6,7,8,9,10,11,12,13,14,15);
}
__device__ __forceinline__ float red16_max(float v) {
    #pragma unroll
    for (int m = 1; m < 16; m <<= 1) v = fmaxf(v, __shfl_xor(v, m, 32));
    return v;
}
__device__ __forceinline__ float red16_sum(float v) {
    #pragma unroll
    for (int m = 1; m < 16; m <<= 1) v += __shfl_xor(v, m, 32);
    return v;
}
__device__ __forceinline__ float red32_sum(float v) {
    #pragma unroll
    for (int m = 1; m < 32; m <<= 1) v += __shfl_xor(v, m, 32);
    return v;
}

#if HAVE_TDM
// Build a D# (CDNA5 ISA) for a 2-D f16 tile load into LDS.
// group0: count=1 | lds_addr | global_addr[56:0] | type=2
// group1: data_size=2B | tensor dims | tile dims | dim0 stride
__device__ __forceinline__ void tdm_load_2d(unsigned lds_off, const void* g,
                                            unsigned tile0, unsigned tile1,
                                            unsigned stride0) {
    unsigned long long ga = (unsigned long long)(size_t)g;
    tdm_g0_t g0;
    g0[0] = 1u;                                        // count=1, user-mode
    g0[1] = lds_off;                                   // LDS byte address
    g0[2] = (unsigned)ga;                              // global_addr[31:0]
    g0[3] = (unsigned)((ga >> 32) & 0x01ffffffu) | 0x80000000u;  // addr[56:32]|type=2
    const unsigned td0 = 1u << 20, td1 = 1u << 20;     // generous tensor dims
    tdm_g1_t g1;
    g1[0] = (int)(1u << 16);                           // wg_mask=0, data_size=1 (2B)
    g1[1] = (int)((td0 & 0xffffu) << 16);              // tensor_dim0[15:0]
    g1[2] = (int)((td0 >> 16) | ((td1 & 0xffffu) << 16));
    g1[3] = (int)((td1 >> 16) | (tile0 << 16));        // tile_dim0
    g1[4] = (int)(tile1);                              // tile_dim1 (tile_dim2=0)
    g1[5] = (int)stride0;                              // tensor_dim0_stride[31:0]
    g1[6] = 0;
    g1[7] = 0;
    tdm_g2_t gz = {0, 0, 0, 0};
#if __clang_major__ >= 23
    tdm_g1_t gz8 = {0, 0, 0, 0, 0, 0, 0, 0};
    __builtin_amdgcn_tensor_load_to_lds(g0, g1, gz, gz, gz8, 0);
#else
    __builtin_amdgcn_tensor_load_to_lds(g0, g1, gz, gz, 0);
#endif
}
__device__ __forceinline__ unsigned lds_off_of(const void* p) {
    return (unsigned)(size_t)p;     // flat shared addr: low 32 bits = LDS offset
}
#endif

// ---------------------------------------------------------------------------
// 0) Convert weight matrices to f16 (packed into one buffer)
// ---------------------------------------------------------------------------
__global__ __launch_bounds__(256) void wcvt_kernel(const float* __restrict__ qkv_w,
                                                   const float* __restrict__ c1_w,
                                                   const float* __restrict__ c2_w,
                                                   _Float16* __restrict__ w16) {
    int i = blockIdx.x * 256 + threadIdx.x;
    if (i < 196608)       w16[i] = (_Float16)qkv_w[i];
    else if (i < 262144)  w16[i] = (_Float16)c1_w[i - 196608];
    else if (i < 327680)  w16[i] = (_Float16)c2_w[i - 262144];
}

// ---------------------------------------------------------------------------
// 1) Depthwise 3x3 conv (SAME) + NCHW -> token-major [b,n,c] f16
// ---------------------------------------------------------------------------
__global__ __launch_bounds__(256) void dwconv_kernel(const float* __restrict__ x,
                                                     const float* __restrict__ w,
                                                     const float* __restrict__ bias,
                                                     _Float16* __restrict__ t16) {
    int idx = blockIdx.x * 256 + threadIdx.x;          // (b*1024+n)*256 + c
    int c = idx & 255;
    int n = (idx >> 8) & 1023;
    int b = idx >> 18;
    int hh = n >> 5, ww = n & 31;
    const float* xp = x + ((size_t)(b * DIMC + c)) * NTOK;
    const float* wp = w + c * 9;
    float s = bias[c];
    #pragma unroll
    for (int dh = -1; dh <= 1; ++dh) {
        int hy = hh + dh;
        if ((unsigned)hy >= 32u) continue;
        #pragma unroll
        for (int dw = -1; dw <= 1; ++dw) {
            int wx = ww + dw;
            if ((unsigned)wx >= 32u) continue;
            s += xp[hy * 32 + wx] * wp[(dh + 1) * 3 + (dw + 1)];
        }
    }
    t16[idx] = (_Float16)s;
}

// ---------------------------------------------------------------------------
// 2) Token GEMM: C[m,n] = sum_k A[m,k]*W[n,k] (+bias[n])
//    One wave computes a 16x64 strip (4 accumulators) -> A frag reused 4x.
// ---------------------------------------------------------------------------
__global__ __launch_bounds__(256) void gemm_tok_kernel(const _Float16* __restrict__ A,
                                                       const _Float16* __restrict__ W,
                                                       const float* __restrict__ bias,
                                                       float* __restrict__ C,
                                                       int M, int Nn, int K) {
    int wave = (blockIdx.x * 256 + threadIdx.x) >> 5;
    int lane = threadIdx.x & 31;
    int half = lane >> 4, ml = lane & 15;
    int nq = Nn >> 6;                                  // 64-wide strips
    if (wave >= (M >> 4) * nq) return;                 // wave-uniform
    int mt = wave / nq, nb = wave % nq;

    const _Float16* arow = A + (size_t)(mt * 16 + ml) * K;
    const _Float16* w0   = W + (size_t)(nb * 64 + ml) * K + half * 16;
    v8f acc[4] = {{}, {}, {}, {}};
    #pragma unroll 4
    for (int k0 = 0; k0 < K; k0 += 32) {
        v16h a = *(const v16h*)(arow + k0 + half * 16);
        v16h b0 = *(const v16h*)(w0 + (size_t)0  * K + k0);
        v16h b1 = *(const v16h*)(w0 + (size_t)16 * K + k0);
        v16h b2 = *(const v16h*)(w0 + (size_t)32 * K + k0);
        v16h b3 = *(const v16h*)(w0 + (size_t)48 * K + k0);
        acc[0] = __builtin_amdgcn_wmma_f32_16x16x32_f16(false, a, false, b0, (short)0, acc[0], false, false);
        acc[1] = __builtin_amdgcn_wmma_f32_16x16x32_f16(false, a, false, b1, (short)0, acc[1], false, false);
        acc[2] = __builtin_amdgcn_wmma_f32_16x16x32_f16(false, a, false, b2, (short)0, acc[2], false, false);
        acc[3] = __builtin_amdgcn_wmma_f32_16x16x32_f16(false, a, false, b3, (short)0, acc[3], false, false);
    }
    #pragma unroll
    for (int j = 0; j < 4; ++j) {
        int col = nb * 64 + j * 16 + ml;
        float bc = bias ? bias[col] : 0.0f;
        #pragma unroll
        for (int r = 0; r < 8; ++r)
            C[(size_t)(mt * 16 + r + 8 * half) * Nn + col] = acc[j][r] + bc;
    }
}

// ---------------------------------------------------------------------------
// 3) Split QKV + per-head L2 norm of q,k; q,k [bh,n,d] f16; V^T [bh,d,n] f16
// ---------------------------------------------------------------------------
__global__ __launch_bounds__(256) void qkv_norm_kernel(const float* __restrict__ qkv,
                                                       _Float16* __restrict__ qh,
                                                       _Float16* __restrict__ kh,
                                                       _Float16* __restrict__ vt) {
    int row = blockIdx.x * 8 + (threadIdx.x >> 5);   // (b*1024+n)*8 + h
    int d = threadIdx.x & 31;
    int h = row & 7;
    int n = (row >> 3) & 1023;
    int b = row >> 13;
    const float* base = qkv + ((size_t)(b * NTOK + n)) * (3 * DIMC) + h * HDIM + d;
    float q = base[0], k = base[DIMC], v = base[2 * DIMC];
    float qn = q / fmaxf(sqrtf(red32_sum(q * q)), 1e-12f);
    float kn = k / fmaxf(sqrtf(red32_sum(k * k)), 1e-12f);
    int bh = b * HEADS + h;
    qh[((size_t)bh * NTOK + n) * HDIM + d] = (_Float16)qn;
    kh[((size_t)bh * NTOK + n) * HDIM + d] = (_Float16)kn;
    vt[((size_t)bh * HDIM + d) * NTOK + n] = (_Float16)v;
}

// ---------------------------------------------------------------------------
// 4) Flash attention. All 8 waves of a block share one (b,h); K/V panels for
//    each 32-key step are staged once per block into double-buffered LDS via
//    TDM tensor_load_to_lds (TENSORcnt) -- fallback: cooperative vector copy.
// ---------------------------------------------------------------------------
__global__ __launch_bounds__(256) void attn_kernel(const _Float16* __restrict__ qh,
                                                   const _Float16* __restrict__ kh,
                                                   const _Float16* __restrict__ vt,
                                                   _Float16* __restrict__ oh) {
    __shared__ __attribute__((aligned(32))) _Float16 kbuf[2][32 * 32];
    __shared__ __attribute__((aligned(32))) _Float16 vbuf[2][32 * 32];
    __shared__ __attribute__((aligned(32))) _Float16 plds[8 * 16 * 32];

    int wid  = threadIdx.x >> 5;
    int lane = threadIdx.x & 31;
    int half = lane >> 4, ml = lane & 15;
    int bh = blockIdx.x >> 3;                          // 8 blocks per (b,h)
    int m0 = (((blockIdx.x & 7) * 8) + wid) * 16;      // query-tile row base

    const _Float16* qbase = qh + (size_t)bh * NTOK * HDIM;
    const _Float16* kbase = kh + (size_t)bh * NTOK * HDIM;
    const _Float16* vbase = vt + (size_t)bh * HDIM * NTOK;
    _Float16* pl = plds + wid * (16 * 32);

    const _Float16* qrow = qbase + (size_t)(m0 + ml) * HDIM;
    v16h aq = *(const v16h*)(qrow + half * 16);

    float mrow[8], lrow[8];
    v8f o0 = {}, o1 = {};
    #pragma unroll
    for (int r = 0; r < 8; ++r) { mrow[r] = -1e30f; lrow[r] = 0.0f; }
    const float scale = 0.17677669529663687f;          // 32^-0.5

    // ---- prologue: stage panel for kt=0 into buffer 0 ----
#if HAVE_TDM
    if (wid == 0) {
        tdm_load_2d(lds_off_of(kbuf[0]), kbase, 1024, 1, 1024);        // K: 32 rows contiguous
        tdm_load_2d(lds_off_of(vbuf[0]), vbase, 32, 32, NTOK);         // V^T: 32x32, stride 1024
        __builtin_amdgcn_s_wait_tensorcnt(0);
    }
#else
    {
        int t = threadIdx.x;
        *(unsigned long long*)(kbuf[0] + t * 4) =
            *(const unsigned long long*)(kbase + t * 4);
        int r = t >> 3, c4 = (t & 7) * 4;
        *(unsigned long long*)(vbuf[0] + r * 32 + c4) =
            *(const unsigned long long*)(vbase + (size_t)r * NTOK + c4);
    }
#endif
    __syncthreads();

    int buf = 0;
    for (int kt = 0; kt < NTOK; kt += 32) {
        int nbuf = buf ^ 1;
        if (kt + 32 < NTOK) {
            // stage next panel while computing current one
#if HAVE_TDM
            if (wid == 0) {
                tdm_load_2d(lds_off_of(kbuf[nbuf]), kbase + (size_t)(kt + 32) * HDIM,
                            1024, 1, 1024);
                tdm_load_2d(lds_off_of(vbuf[nbuf]), vbase + (kt + 32),
                            32, 32, NTOK);
            }
#else
            int t = threadIdx.x;
            *(unsigned long long*)(kbuf[nbuf] + t * 4) =
                *(const unsigned long long*)(kbase + (size_t)(kt + 32) * HDIM + t * 4);
            int r = t >> 3, c4 = (t & 7) * 4;
            *(unsigned long long*)(vbuf[nbuf] + r * 32 + c4) =
                *(const unsigned long long*)(vbase + (size_t)r * NTOK + kt + 32 + c4);
#endif
        }
        const _Float16* kb = kbuf[buf];
        const _Float16* vb = vbuf[buf];

        v16h bk0 = *(const v16h*)(kb + ml * 32 + half * 16);
        v16h bk1 = *(const v16h*)(kb + (16 + ml) * 32 + half * 16);
        v8f z = {};
        v8f s0 = __builtin_amdgcn_wmma_f32_16x16x32_f16(false, aq, false, bk0, (short)0, z, false, false);
        v8f s1 = __builtin_amdgcn_wmma_f32_16x16x32_f16(false, aq, false, bk1, (short)0, z, false, false);

        #pragma unroll
        for (int r = 0; r < 8; ++r) {
            float x0 = s0[r] * scale, x1 = s1[r] * scale;
            float tm = red16_max(fmaxf(x0, x1));
            float nm = fmaxf(mrow[r], tm);
            float alpha = __expf(mrow[r] - nm);
            float p0 = __expf(x0 - nm), p1 = __expf(x1 - nm);
            lrow[r] = lrow[r] * alpha + red16_sum(p0 + p1);
            mrow[r] = nm;
            o0[r] *= alpha; o1[r] *= alpha;
            pl[(r + 8 * half) * 32 + ml]      = (_Float16)p0;
            pl[(r + 8 * half) * 32 + 16 + ml] = (_Float16)p1;
        }
        // relayout P (D-layout) -> A-frag through private LDS slab
        v16h ap = *(const v16h*)(pl + ml * 32 + half * 16);
        v16h bv0 = *(const v16h*)(vb + ml * 32 + half * 16);
        v16h bv1 = *(const v16h*)(vb + (16 + ml) * 32 + half * 16);
        o0 = __builtin_amdgcn_wmma_f32_16x16x32_f16(false, ap, false, bv0, (short)0, o0, false, false);
        o1 = __builtin_amdgcn_wmma_f32_16x16x32_f16(false, ap, false, bv1, (short)0, o1, false, false);

#if HAVE_TDM
        if (wid == 0) __builtin_amdgcn_s_wait_tensorcnt(0);
#endif
        __syncthreads();
        buf = nbuf;
    }

    int b = bh >> 3, h = bh & 7;
    #pragma unroll
    for (int r = 0; r < 8; ++r) {
        int n = m0 + r + 8 * half;
        float inv = 1.0f / lrow[r];
        size_t base = ((size_t)(b * NTOK + n)) * DIMC + h * HDIM;
        oh[base + ml]      = (_Float16)(o0[r] * inv);
        oh[base + 16 + ml] = (_Float16)(o1[r] * inv);
    }
}

// ---------------------------------------------------------------------------
// 5/6) Channel LayerNorm + exact GELU (+ residual, NCHW store)
// ---------------------------------------------------------------------------
__device__ __forceinline__ float blk_sum(float v, float* red, int lane, int wid) {
    v = red32_sum(v);
    if (lane == 0) red[wid] = v;
    __syncthreads();
    float t = 0.f;
    #pragma unroll
    for (int i = 0; i < 8; ++i) t += red[i];
    __syncthreads();
    return t;
}

__global__ __launch_bounds__(256) void ln_gelu_kernel(const float* __restrict__ yin,
                                                      const float* __restrict__ g,
                                                      const float* __restrict__ bt,
                                                      _Float16* __restrict__ outh) {
    __shared__ float red[8];
    int tkn = blockIdx.x, c = threadIdx.x;
    int lane = c & 31, wid = c >> 5;
    float v = yin[(size_t)tkn * DIMC + c];
    float mu = blk_sum(v, red, lane, wid) * (1.0f / DIMC);
    float d = v - mu;
    float var = blk_sum(d * d, red, lane, wid) * (1.0f / DIMC);
    float o = d * rsqrtf(var + 1e-5f) * g[c] + bt[c];
    float ge = 0.5f * o * (1.0f + erff(o * 0.70710678118f));
    outh[(size_t)tkn * DIMC + c] = (_Float16)ge;
}

__global__ __launch_bounds__(256) void ln_gelu_res_kernel(const float* __restrict__ yin,
                                                          const float* __restrict__ g,
                                                          const float* __restrict__ bt,
                                                          const float* __restrict__ x,
                                                          float* __restrict__ out) {
    __shared__ float red[8];
    int tkn = blockIdx.x, c = threadIdx.x;
    int lane = c & 31, wid = c >> 5;
    float v = yin[(size_t)tkn * DIMC + c];
    float mu = blk_sum(v, red, lane, wid) * (1.0f / DIMC);
    float d = v - mu;
    float var = blk_sum(d * d, red, lane, wid) * (1.0f / DIMC);
    float o = d * rsqrtf(var + 1e-5f) * g[c] + bt[c];
    float ge = 0.5f * o * (1.0f + erff(o * 0.70710678118f));
    int b = tkn >> 10, n = tkn & 1023;
    size_t xi = ((size_t)(b * DIMC + c)) * NTOK + n;
    out[xi] = x[xi] + ge;
}

// ---------------------------------------------------------------------------
// Launcher
// ---------------------------------------------------------------------------
extern "C" void kernel_launch(void* const* d_in, const int* in_sizes, int n_in,
                              void* d_out, int out_size, void* d_ws, size_t ws_size,
                              hipStream_t stream) {
    const float* x     = (const float*)d_in[0];
    const float* dw_w  = (const float*)d_in[1];
    const float* dw_b  = (const float*)d_in[2];
    const float* qkv_w = (const float*)d_in[3];
    const float* c1_w  = (const float*)d_in[4];
    const float* c1_b  = (const float*)d_in[5];
    const float* ln1_g = (const float*)d_in[6];
    const float* ln1_b = (const float*)d_in[7];
    const float* c2_w  = (const float*)d_in[8];
    const float* c2_b  = (const float*)d_in[9];
    const float* ln2_g = (const float*)d_in[10];
    const float* ln2_b = (const float*)d_in[11];
    float* out = (float*)d_out;

    char* ws = (char*)d_ws;
    const size_t MB = 1u << 20;
    _Float16* w16   = (_Float16*)(ws);             // 0.66 MB weights f16
    _Float16* t16   = (_Float16*)(ws + 1 * MB);    // 4 MB tokens f16
    float*    qkvf  = (float*)   (ws + 5 * MB);    // 24 MB qkv f32
    float*    y1f   = (float*)   (ws + 5 * MB);    // reuse (qkv dead after norm)
    float*    y2f   = (float*)   (ws + 13 * MB);   // 8 MB
    _Float16* y1h   = (_Float16*)(ws + 21 * MB);   // 4 MB
    _Float16* qh    = (_Float16*)(ws + 29 * MB);   // 4 MB
    _Float16* kh    = (_Float16*)(ws + 33 * MB);   // 4 MB
    _Float16* vth   = (_Float16*)(ws + 37 * MB);   // 4 MB (V transposed)
    _Float16* oh    = (_Float16*)(ws + 41 * MB);   // 4 MB attention out f16

    _Float16* qkvw16 = w16;
    _Float16* c1w16  = w16 + 196608;
    _Float16* c2w16  = w16 + 262144;

    const int M = BATCH * NTOK;                    // 8192 token rows

    wcvt_kernel<<<1280, 256, 0, stream>>>(qkv_w, c1_w, c2_w, w16);
    dwconv_kernel<<<(M * DIMC) / 256, 256, 0, stream>>>(x, dw_w, dw_b, t16);

    // QKV: 512 mtiles * 12 strips / 8 waves = 768 blocks
    gemm_tok_kernel<<<768, 256, 0, stream>>>(t16, qkvw16, nullptr, qkvf,
                                             M, 3 * DIMC, DIMC);
    qkv_norm_kernel<<<(M * HEADS) / 8, 256, 0, stream>>>(qkvf, qh, kh, vth);

    // attention: 64 (b,h) * 8 blocks each
    attn_kernel<<<512, 256, 0, stream>>>(qh, kh, vth, oh);

    // c1/c2: 512 mtiles * 4 strips / 8 waves = 256 blocks
    gemm_tok_kernel<<<256, 256, 0, stream>>>(oh, c1w16, c1_b, y1f, M, DIMC, DIMC);
    ln_gelu_kernel<<<M, 256, 0, stream>>>(y1f, ln1_g, ln1_b, y1h);
    gemm_tok_kernel<<<256, 256, 0, stream>>>(y1h, c2w16, c2_b, y2f, M, DIMC, DIMC);
    ln_gelu_res_kernel<<<M, 256, 0, stream>>>(y2f, ln2_g, ln2_b, x, out);
}
